// Info_Dropout_18391049961713
// MI455X (gfx1250) — matrix operand order repaired and checked by reading, not compiled, not run on hardware
//
#include <hip/hip_runtime.h>
#include <hip/hip_bf16.h>
#include <stdint.h>

typedef __attribute__((ext_vector_type(16))) _Float16 v16h;
typedef __attribute__((ext_vector_type(8)))  float    v8f;

#define B_      32
#define NPIX    4096        // 64*64
#define NPATCH  9
#define S_      5120        // int(1.25 * 4096)
#define NROWS   2048        // 32 * 64
#define NHALF_  5242880u    // (2048*5120)/2
#define DCOUNT  1179648.0f  // 32*9*64*64

// Stand-in for np.random.default_rng(42).integers(-2, 3, 9) (fixed constants).
__constant__ int c_si[NPATCH] = { 1,-2, 2, 0,-1, 2,-2, 1, 0};
__constant__ int c_sj[NPATCH] = {-1, 2, 0,-2, 1, 1, 2,-2, 0};

__device__ __forceinline__ int refl(int i) {          // jnp reflect (no edge repeat)
    return i < 0 ? -i : (i > 63 ? 126 - i : i);
}

__global__ void k_init(float* acc) {
    if (threadIdx.x < 64) acc[threadIdx.x] = 0.0f;
}

// One workgroup per (patch k, batch b): SSD map (66x66) in LDS, then
// 3x3 valid box filter as  Final(64x64) = Wv(64x96) * [ D(80x96) * Wh(96x64) ]
// using v_wmma_f32_16x16x32_f16, K padded with zeros.
__global__ void k_dist(const float* __restrict__ xold,
                       float* __restrict__ dist,
                       float* __restrict__ acc) {
    __shared__ _Float16 dh[80][96];   // padded SSD map (f16)
    __shared__ _Float16 th[96][64];   // horizontal-pass result
    __shared__ float    rs[256];

    const int k = blockIdx.x, b = blockIdx.y;
    const int tid = threadIdx.x;
    const int lane = tid & 31, wave = tid >> 5;
    const int hs = (lane >= 16) ? 1 : 0;   // upper half-wave
    const int mn = lane & 15;

    for (int i = tid; i < 80 * 96; i += 256) (&dh[0][0])[i] = (_Float16)0.0f;
    for (int i = tid; i < 96 * 64; i += 256) (&th[0][0])[i] = (_Float16)0.0f;
    __syncthreads();

    const int si = c_si[k], sj = c_sj[k];
    const float* xb = xold + (size_t)b * 3 * NPIX;
    for (int p = tid; p < 66 * 66; p += 256) {
        const int y = p / 66, x = p % 66;
        const int ub = refl(y - 1), vb = refl(x - 1);
        const int us = refl(y - 1 + si), vs = refl(x - 1 + sj);
        float s = 0.0f;
        #pragma unroll
        for (int c = 0; c < 3; ++c) {
            const float a = xb[c * NPIX + ub * 64 + vb];
            const float bsh = xb[c * NPIX + us * 64 + vs];
            const float d = a - bsh;
            s += d * d;
        }
        dh[y][x] = (_Float16)s;
    }
    __syncthreads();

    // Pass 1: T = D * Wh   (Wh[j][x] = 1 iff x <= j <= x+2). 5x4 tiles, K = 3x32.
    for (int t = wave; t < 20; t += 8) {
        const int tr = t >> 2, tc = t & 3;
        v8f cacc = {};
        #pragma unroll
        for (int ks = 0; ks < 3; ++ks) {
            v16h a, wb;
            #pragma unroll
            for (int h = 0; h < 16; ++h) {
                const int v = h >> 1, par = h & 1;
                const int Ka = ((v >= 4) ? 16 : 0) + (hs ? 8 : 0) + ((v & 3) << 1) + par;
                a[h] = dh[tr * 16 + mn][ks * 32 + Ka];
                const int j = ks * 32 + Ka;
                const int x = tc * 16 + mn;
                wb[h] = (j >= x && j <= x + 2) ? (_Float16)1.0f : (_Float16)0.0f;
            }
            cacc = __builtin_amdgcn_wmma_f32_16x16x32_f16(false, a, false, wb,
                                                          (short)0, cacc, false, false);
        }
        #pragma unroll
        for (int r = 0; r < 8; ++r)
            th[tr * 16 + r + (hs ? 8 : 0)][tc * 16 + mn] = (_Float16)cacc[r];
    }
    __syncthreads();

    // Pass 2: Final = Wv * T   (Wv[m][j] = 1 iff m <= j <= m+2). 4x4 tiles.
    float lsum = 0.0f;
    float* drow = dist + (size_t)(b * NPATCH + k) * NPIX;
    for (int t = wave; t < 16; t += 8) {
        const int tr = t >> 2, tc = t & 3;
        v8f cacc = {};
        #pragma unroll
        for (int ks = 0; ks < 3; ++ks) {
            v16h a, tb;
            #pragma unroll
            for (int h = 0; h < 16; ++h) {
                const int v = h >> 1, par = h & 1;
                const int Ka = ((v >= 4) ? 16 : 0) + (hs ? 8 : 0) + ((v & 3) << 1) + par;
                const int j = ks * 32 + Ka;
                const int m = tr * 16 + mn;
                a[h] = (j >= m && j <= m + 2) ? (_Float16)1.0f : (_Float16)0.0f;
                tb[h] = th[ks * 32 + Ka][tc * 16 + mn];
            }
            cacc = __builtin_amdgcn_wmma_f32_16x16x32_f16(false, a, false, tb,
                                                          (short)0, cacc, false, false);
        }
        #pragma unroll
        for (int r = 0; r < 8; ++r) {
            const int y = tr * 16 + r + (hs ? 8 : 0);
            const int x = tc * 16 + mn;
            const float vv = cacc[r];
            drow[y * 64 + x] = vv;
            lsum += vv;
        }
    }

    rs[tid] = lsum;
    __syncthreads();
    for (int o = 128; o > 0; o >>= 1) {
        if (tid < o) rs[tid] += rs[tid + o];
        __syncthreads();
    }
    if (tid == 0) atomicAdd(&acc[0], rs[0]);
}

__global__ void k_prob(const float* __restrict__ dist, float* __restrict__ acc,
                       float* __restrict__ prob) {
    __shared__ float rs[256];
    const int b = blockIdx.x, tid = threadIdx.x;
    const float mean = acc[0] * (1.0f / DCOUNT);
    const float sc = -1.0f / (2.0f * mean);
    float lsum = 0.0f;
    for (int i = tid; i < NPIX; i += 256) {
        float s = 0.0f;
        #pragma unroll
        for (int k = 0; k < NPATCH; ++k)
            s += __expf(dist[(size_t)(b * NPATCH + k) * NPIX + i] * sc);
        const float p = __powf(s * (1.0f / 9.0f), 33.333332f);   // ^(1/TEMP)
        prob[b * NPIX + i] = p;
        lsum += p;
    }
    rs[tid] = lsum;
    __syncthreads();
    for (int o = 128; o > 0; o >>= 1) {
        if (tid < o) rs[tid] += rs[tid + o];
        __syncthreads();
    }
    if (tid == 0) atomicAdd(&acc[1 + b], rs[0]);
}

// Per-batch inclusive scan of weights -> CDF (N = 4096): 16/thread + LDS scan.
__global__ void k_cdf(const float* __restrict__ prob, const float* __restrict__ acc,
                      float* __restrict__ cdf) {
    __shared__ float ssum[256];
    const int b = blockIdx.x, tid = threadIdx.x;
    const float inv = 1.0f / (acc[1 + b] + 1e-16f);
    const int base = tid * 16;
    float v[16];
    float run = 0.0f;
    #pragma unroll
    for (int j = 0; j < 16; ++j) {
        run += prob[b * NPIX + base + j] * inv + 1e-8f;
        v[j] = run;
    }
    ssum[tid] = run;
    __syncthreads();
    for (int o = 1; o < 256; o <<= 1) {
        const float add = (tid >= o) ? ssum[tid - o] : 0.0f;
        __syncthreads();
        ssum[tid] += add;
        __syncthreads();
    }
    const float off = ssum[tid] - run;
    #pragma unroll
    for (int j = 0; j < 16; ++j) cdf[b * NPIX + base + j] = v[j] + off;
}

__global__ void k_copy(const float4* __restrict__ x, float4* __restrict__ out, int n4) {
    const int i = blockIdx.x * 256 + threadIdx.x;
    if (i < n4) out[i] = x[i];
}

__device__ __forceinline__ uint32_t rotl32(uint32_t v, int d) {
    return (v << d) | (v >> (32 - d));
}
// JAX threefry2x32, key = (0, 42)  (jax.random.key(42)).
__device__ __forceinline__ void threefry2x32(uint32_t c0, uint32_t c1,
                                             uint32_t& o0, uint32_t& o1) {
    const uint32_t k0 = 0u, k1 = 42u, k2 = 0u ^ 42u ^ 0x1BD11BDAu;
    const uint32_t ks[3] = {k0, k1, k2};
    uint32_t x0 = c0 + k0, x1 = c1 + k1;
    const int ra[4] = {13, 15, 26, 6}, rb[4] = {17, 29, 16, 24};
    #pragma unroll
    for (int g = 0; g < 5; ++g) {
        const int* r = (g & 1) ? rb : ra;
        #pragma unroll
        for (int i = 0; i < 4; ++i) {
            x0 += x1; x1 = rotl32(x1, r[i]); x1 ^= x0;
        }
        x0 += ks[(g + 1) % 3];
        x1 += ks[(g + 2) % 3] + (uint32_t)(g + 1);
    }
    o0 = x0; o1 = x1;
}

// One workgroup per (b,c) row: stage CDF row into LDS via async-to-LDS DMA,
// draw S uniforms, inverse-CDF (searchsorted 'right'), scatter zeros into out.
__global__ void k_sample(const int* __restrict__ flag, const float* __restrict__ cdf,
                         float* __restrict__ out) {
    if (flag[0] == 0) return;                       // grid-uniform
    __shared__ float cs[NPIX];
    const int r = blockIdx.x, tid = threadIdx.x;
    const int b = r >> 6;
    const float* src = cdf + (size_t)b * NPIX;

    #pragma unroll
    for (int i = 0; i < 4; ++i) {                   // 256 thr x 4 x 16B = 16 KB
        const int e = (tid + i * 256) * 4;
        const uint32_t lds = (uint32_t)(size_t)(const void*)&cs[e];
        const unsigned long long ga = (unsigned long long)(const void*)(src + e);
        asm volatile("global_load_async_to_lds_b128 %0, %1, off"
                     :: "v"(lds), "v"(ga) : "memory");
    }
    asm volatile("s_wait_asynccnt 0x0" ::: "memory");
    __syncthreads();

    const float total = cs[NPIX - 1];
    const size_t rowbase = (size_t)r * NPIX;
    #pragma unroll 4
    for (int it = 0; it < 20; ++it) {               // 20 * 256 = 5120 samples
        const uint32_t s = (uint32_t)(tid + it * 256);
        const uint32_t e = (uint32_t)r * (uint32_t)S_ + s;
        uint32_t o0, o1, bits;
        if (e < NHALF_) { threefry2x32(e, e + NHALF_, o0, o1); bits = o0; }
        else            { threefry2x32(e - NHALF_, e, o0, o1); bits = o1; }
        const uint32_t fb = (bits >> 9) | 0x3f800000u;
        float u;
        __builtin_memcpy(&u, &fb, 4);
        u = (u - 1.0f) * total;                     // uniform [0,1) * cdf[-1]
        int lo = 0, hi = NPIX;
        while (lo < hi) {                           // searchsorted(side='right')
            const int mid = (lo + hi) >> 1;
            if (cs[mid] <= u) lo = mid + 1; else hi = mid;
        }
        const int idx = lo > NPIX - 1 ? NPIX - 1 : lo;
        out[rowbase + idx] = 0.0f;                  // mask==0 -> x*0
    }
}

extern "C" void kernel_launch(void* const* d_in, const int* in_sizes, int n_in,
                              void* d_out, int out_size, void* d_ws, size_t ws_size,
                              hipStream_t stream) {
    (void)in_sizes; (void)n_in; (void)out_size; (void)ws_size;
    const float* x_old = (const float*)d_in[0];   // (32,3,64,64) f32
    const float* x     = (const float*)d_in[1];   // (32,64,64,64) f32
    const int*   flag  = (const int*)d_in[2];     // use_info_drop
    float* out = (float*)d_out;

    float* acc  = (float*)d_ws;        // [0]=global dist sum, [1..32]=batch sums
    float* dist = acc + 64;            // 32*9*4096 floats
    float* prob = dist + 32 * NPATCH * NPIX;   // 32*4096
    float* cdf  = prob + 32 * NPIX;            // 32*4096

    k_init<<<1, 64, 0, stream>>>(acc);
    k_dist<<<dim3(NPATCH, B_), 256, 0, stream>>>(x_old, dist, acc);
    k_prob<<<B_, 256, 0, stream>>>(dist, acc, prob);
    k_cdf<<<B_, 256, 0, stream>>>(prob, acc, cdf);
    k_copy<<<8192, 256, 0, stream>>>((const float4*)x, (float4*)out, 2097152);
    k_sample<<<NROWS, 256, 0, stream>>>(flag, cdf, out);
}
